// RoPEAttentionLayer_6751688590009
// MI455X (gfx1250) — matrix operand order repaired and checked by reading, not compiled, not run on hardware
//
#include <hip/hip_runtime.h>

// ---------------------------------------------------------------------------
// Types for CDNA5 WMMA (wave32)
// ---------------------------------------------------------------------------
typedef __bf16 bf16;
typedef __attribute__((ext_vector_type(16))) __bf16 v16bf;
typedef __attribute__((ext_vector_type(8)))  __bf16 v8bf;
typedef __attribute__((ext_vector_type(4)))  __bf16 v4bf;
typedef __attribute__((ext_vector_type(8)))  float  v8f;
typedef __attribute__((ext_vector_type(4)))  unsigned int v4u;
typedef __attribute__((ext_vector_type(8)))  int    v8i;
typedef __attribute__((ext_vector_type(4)))  int    v4i;

#if __has_builtin(__builtin_amdgcn_tensor_load_to_lds)
#define HAVE_TDM 1
#else
#define HAVE_TDM 0
#endif

__device__ __forceinline__ v8f wmma_bf16(v16bf a, v16bf b, v8f c) {
  // D = A(16x32 bf16) * B(32x16 bf16) + C(16x16 f32)
  return __builtin_amdgcn_wmma_f32_16x16x32_bf16(
      /*neg_a=*/false, a, /*neg_b=*/false, b,
      /*c_mod=*/(short)0, c, /*reuse_a=*/false, /*reuse_b=*/false);
}

__device__ __forceinline__ v8f zero8() {
  v8f c;
#pragma unroll
  for (int i = 0; i < 8; ++i) c[i] = 0.0f;
  return c;
}

__device__ __forceinline__ bf16 to_bf16(float f) { return (bf16)f; }

// ---------------------------------------------------------------------------
// CDNA5 async global->LDS copy (ASYNCcnt path, CDNA5 ISA).
// LDS operand = low 32 bits of the generic pointer (LDS aperture rule).
// ---------------------------------------------------------------------------
__device__ __forceinline__ void async_load_b128(const void* gptr, void* lptr) {
  unsigned int laddr = (unsigned int)(unsigned long long)(uintptr_t)lptr;
  asm volatile("global_load_async_to_lds_b128 %0, %1, off"
               :: "v"(laddr), "v"((unsigned long long)(uintptr_t)gptr)
               : "memory");
}

__device__ __forceinline__ void wait_async0() {
#if __has_builtin(__builtin_amdgcn_s_wait_asynccnt)
  __builtin_amdgcn_s_wait_asynccnt(0);
#else
  asm volatile("s_wait_asynccnt 0" ::: "memory");
#endif
}

// ---------------------------------------------------------------------------
// Tensor Data Mover: 2D bf16 tile load, global -> LDS, with hardware row
// padding (16B pad every 64 elements => LDS row stride 72 halves).
// Descriptor layout per CDNA5 ISA.
// This toolchain's builtin is the 6-arg (clang-23 / therock) form:
//   (uint32x4 g0, int32x8 g1, int32x4 g2, int32x4 g3, int32x8 gx, i32 cpol)
// ---------------------------------------------------------------------------
__device__ __forceinline__ void tdm_load_2d_bf16(
    const bf16* gptr, bf16* lptr,
    unsigned tdim0, unsigned tdim1, unsigned stride0,
    unsigned tile0, unsigned tile1)
{
#if HAVE_TDM
  unsigned long long ga = (unsigned long long)(uintptr_t)gptr;
  unsigned int la = (unsigned int)(uintptr_t)lptr;
  v4u g0;
  g0[0] = 1u;                                        // count=1 (valid)
  g0[1] = la;                                        // lds_addr
  g0[2] = (unsigned)(ga & 0xffffffffu);              // global_addr[31:0]
  g0[3] = (unsigned)((ga >> 32) & 0x1ffffffu)        // global_addr[56:32]
        | (2u << 30);                                // type=2 ("image")
  v8i g1;
  g1[0] = (int)((1u << 16)        // data_size = 2B
              | (1u << 20)        // pad_enable
              | (4u << 22)        // pad_interval: 32 DWORDs (=64 bf16)
              | (3u << 25));      // pad_amount: 4 DWORDs (=16B)
  g1[1] = (int)((tdim0 & 0xffffu) << 16);                       // tensor_dim0[15:0]
  g1[2] = (int)((tdim0 >> 16) | ((tdim1 & 0xffffu) << 16));     // d0 hi | d1 lo
  g1[3] = (int)((tdim1 >> 16) | (tile0 << 16));                 // d1 hi | tile_dim0
  g1[4] = (int)(tile1 & 0xffffu);                               // tile_dim1 (dim2=0)
  g1[5] = (int)stride0;                                         // dim0_stride[31:0]
  g1[6] = 0;                                                    // stride hi / d1stride
  g1[7] = 0;
  v4i z4; z4[0] = 0; z4[1] = 0; z4[2] = 0; z4[3] = 0;
  v8i z8;
#pragma unroll
  for (int i = 0; i < 8; ++i) z8[i] = 0;
  __builtin_amdgcn_tensor_load_to_lds(g0, g1, z4, z4, z8, 0);
#else
  (void)gptr; (void)lptr; (void)tdim0; (void)tdim1;
  (void)stride0; (void)tile0; (void)tile1;
#endif
}

__device__ __forceinline__ void wait_tensor0() {
#if __has_builtin(__builtin_amdgcn_s_wait_tensorcnt)
  __builtin_amdgcn_s_wait_tensorcnt(0);
#else
  asm volatile("s_wait_tensorcnt 0" ::: "memory");
#endif
}

// ---------------------------------------------------------------------------
// fp32 -> bf16 conversion (4-wide)
// ---------------------------------------------------------------------------
__global__ void k_cvt(const float* __restrict__ in, bf16* __restrict__ out, int n4) {
  int i = blockIdx.x * 256 + threadIdx.x;
  if (i < n4) {
    float4 f = ((const float4*)in)[i];
    v4bf o;
    o[0] = to_bf16(f.x); o[1] = to_bf16(f.y);
    o[2] = to_bf16(f.z); o[3] = to_bf16(f.w);
    ((v4bf*)out)[i] = o;
  }
}

// ---------------------------------------------------------------------------
// GEMM: out[M,N] = A[M,K](bf16) * W[N,K]^T(bf16)  (+bias)(+resid)(+relu)
// 256 threads = 8 waves; block tile 64x128; wave tile 32x32; K step 64.
// A/B panels double-buffered in LDS, staged by the Tensor Data Mover
// (wave 0 issues 2 descriptors/stage), padded rows (stride 72) for banks.
// ---------------------------------------------------------------------------
#define LDA_L 72   // padded LDS row stride in bf16 elements (64 + 8)

__global__ __launch_bounds__(256) void k_gemm(
    const bf16* __restrict__ A, const bf16* __restrict__ W,
    const float* __restrict__ bias, const float* __restrict__ resid,
    float* __restrict__ outF, bf16* __restrict__ outB,
    int M, int N, int K, int relu)
{
  const int tid  = threadIdx.x;
  const int lane = tid & 31;
  const int wave = tid >> 5;
  const int half = lane >> 4;   // 0..1
  const int l16  = lane & 15;   // 0..15
  const int wm   = wave >> 2;   // 0..1
  const int wn   = wave & 3;    // 0..3
  const int bm   = blockIdx.y * 64;
  const int bn   = blockIdx.x * 128;

  __shared__ alignas(16) bf16 lA[2][64 * LDA_L];    // 9 KB per buffer
  __shared__ alignas(16) bf16 lB[2][128 * LDA_L];   // 18 KB per buffer

  auto stage = [&](int buf, int kc) {
#if HAVE_TDM
    if (wave == 0) {
      tdm_load_2d_bf16(A + (size_t)bm * K + kc, &lA[buf][0],
                       (unsigned)K, (unsigned)M, (unsigned)K, 64, 64);
      tdm_load_2d_bf16(W + (size_t)bn * K + kc, &lB[buf][0],
                       (unsigned)K, (unsigned)N, (unsigned)K, 64, 128);
    }
#else
    // fallback: per-lane async copies (64x64 A = 512 chunks, 128x64 B = 1024)
#pragma unroll
    for (int j = 0; j < 2; ++j) {
      int idx = tid + j * 256;
      int row = idx >> 3, seg = idx & 7;
      async_load_b128(A + (size_t)(bm + row) * K + kc + seg * 8,
                      &lA[buf][row * LDA_L + seg * 8]);
    }
#pragma unroll
    for (int j = 0; j < 4; ++j) {
      int idx = tid + j * 256;
      int row = idx >> 3, seg = idx & 7;
      async_load_b128(W + (size_t)(bn + row) * K + kc + seg * 8,
                      &lB[buf][row * LDA_L + seg * 8]);
    }
#endif
  };

  stage(0, 0);

  v8f c00 = zero8(), c01 = zero8(), c10 = zero8(), c11 = zero8();

  for (int kc = 0; kc < K; kc += 64) {
    const int buf = (kc >> 6) & 1;
#if HAVE_TDM
    if (wave == 0) wait_tensor0();
#else
    wait_async0();
#endif
    __syncthreads();                  // current buffer fully resident
    if (kc + 64 < K) stage(buf ^ 1, kc + 64);   // prefetch next while computing

#pragma unroll
    for (int kk = 0; kk < 64; kk += 32) {
      // A operands: lane l16 = row; K-chunks [half*8,+8) and [16+half*8,+8)
      union { v16bf v; v8bf h[2]; } a0, a1, b0, b1;
      const int ma = wm * 32 + l16;
      a0.h[0] = *(const v8bf*)&lA[buf][(ma)      * LDA_L + kk + half * 8];
      a0.h[1] = *(const v8bf*)&lA[buf][(ma)      * LDA_L + kk + 16 + half * 8];
      a1.h[0] = *(const v8bf*)&lA[buf][(ma + 16) * LDA_L + kk + half * 8];
      a1.h[1] = *(const v8bf*)&lA[buf][(ma + 16) * LDA_L + kk + 16 + half * 8];
      // B operands: lane l16 = column; K-chunks [half*8,+8) and [16+half*8,+8)
      const int nb = wn * 32 + l16;
      b0.h[0] = *(const v8bf*)&lB[buf][(nb)      * LDA_L + kk + half * 8];
      b0.h[1] = *(const v8bf*)&lB[buf][(nb)      * LDA_L + kk + 16 + half * 8];
      b1.h[0] = *(const v8bf*)&lB[buf][(nb + 16) * LDA_L + kk + half * 8];
      b1.h[1] = *(const v8bf*)&lB[buf][(nb + 16) * LDA_L + kk + 16 + half * 8];

      c00 = wmma_bf16(a0.v, b0.v, c00);
      c01 = wmma_bf16(a0.v, b1.v, c01);
      c10 = wmma_bf16(a1.v, b0.v, c10);
      c11 = wmma_bf16(a1.v, b1.v, c11);
    }
  }

  // Epilogue. C layout: VGPR r -> row (r + 8*half), column l16.
  const int m0 = bm + wm * 32;
  const int n0 = bn + wn * 32;
  auto store_tile = [&](v8f c, int mt, int nt) {
#pragma unroll
    for (int r = 0; r < 8; ++r) {
      int row = mt + r + half * 8;
      int col = nt + l16;
      float v = c[r];
      if (bias)  v += bias[col];
      if (resid) v += resid[(size_t)row * N + col];
      if (relu)  v = v > 0.0f ? v : 0.0f;
      if (outF)  outF[(size_t)row * N + col] = v;
      if (outB)  outB[(size_t)row * N + col] = to_bf16(v);
    }
  };
  store_tile(c00, m0,      n0);
  store_tile(c01, m0,      n0 + 16);
  store_tile(c10, m0 + 16, n0);
  store_tile(c11, m0 + 16, n0 + 16);
}

// ---------------------------------------------------------------------------
// RoPE: in fp32 [B,S,H,64] -> out bf16 [B,H,S,64]
// ---------------------------------------------------------------------------
__global__ void k_rope(const float* __restrict__ in, bf16* __restrict__ out,
                       int BSH, int S, int H)
{
  int idx = blockIdx.x * 256 + threadIdx.x;   // over BSH*32
  if (idx >= BSH * 32) return;
  int i = idx & 31;
  int t = idx >> 5;          // (b*S + s)*H + h
  int h = t % H; t /= H;
  int s = t % S;
  int b = t / S;
  const float* row = in + ((size_t)((b * S + s) * H + h)) * 64;
  float x1 = row[i], x2 = row[i + 32];
  float inv = __expf(-((float)(2 * i) / 64.0f) * 9.210340371976184f);
  float ang = (float)s * inv;
  float sn, cs;
  __sincosf(ang, &sn, &cs);
  bf16* orow = out + ((size_t)((b * H + h) * S + s)) * 64;
  orow[i]      = to_bf16(x1 * cs - x2 * sn);
  orow[i + 32] = to_bf16(x2 * cs + x1 * sn);
}

// ---------------------------------------------------------------------------
// V transpose: vb bf16 [B*S, 512] -> vtb bf16 [B*H, 64, S]
// ---------------------------------------------------------------------------
__global__ void k_transpose_v(const bf16* __restrict__ vb, bf16* __restrict__ vtb,
                              int S, int H)
{
  size_t idx = (size_t)blockIdx.x * 256 + threadIdx.x;  // output-linear
  int s = (int)(idx % S);
  size_t t = idx / S;
  int d = (int)(t % 64); t /= 64;
  int h = (int)(t % H);
  int b = (int)(t / H);
  vtb[idx] = vb[((size_t)(b * S + s)) * 512 + h * 64 + d];
}

// ---------------------------------------------------------------------------
// Flash attention: Q,K bf16 [B*H, S, 64]; Vt bf16 [B*H, 64, S];
// ctx bf16 [B, S, 512]. 4 waves/WG, 16 queries/wave, 32-key tiles.
// K/V tiles shared across waves -> double-buffered async LDS staging,
// padded LDS rows (72 / 40 halves) against bank conflicts.
// ---------------------------------------------------------------------------
#define LDK_L 72   // K tile row stride (64 + 8)
#define LDV_L 40   // V^T tile row stride (32 + 8)

__global__ __launch_bounds__(128) void k_flash(
    const bf16* __restrict__ Q, const bf16* __restrict__ Kt,
    const bf16* __restrict__ Vt, bf16* __restrict__ ctx, int S, int H)
{
  const int tid  = threadIdx.x;
  const int lane = tid & 31;
  const int wave = tid >> 5;
  const int half = lane >> 4;
  const int l16  = lane & 15;
  const int bh   = blockIdx.y;          // b*H + h
  const int b    = bh / H, h = bh % H;
  const int q0   = blockIdx.x * 64 + wave * 16;
  const float scale = 0.125f;           // 1/sqrt(64)

  __shared__ alignas(16) bf16 kbuf[2][32 * LDK_L];
  __shared__ alignas(16) bf16 vbuf[2][64 * LDV_L];
  __shared__ alignas(16) bf16 pbuf[4][16][32];

  auto stage = [&](int buf, int kt) {
    // K tile 32x64 = 256 x 16B chunks (2/thread)
#pragma unroll
    for (int j = 0; j < 2; ++j) {
      int idx = tid + j * 128;
      int row = idx >> 3, seg = idx & 7;
      async_load_b128(Kt + ((size_t)bh * S + kt + row) * 64 + seg * 8,
                      &kbuf[buf][row * LDK_L + seg * 8]);
    }
    // V^T tile 64x32 = 256 x 16B chunks (2/thread)
#pragma unroll
    for (int j = 0; j < 2; ++j) {
      int idx = tid + j * 128;
      int row = idx >> 2, seg = idx & 3;
      async_load_b128(Vt + ((size_t)bh * 64 + row) * S + kt + seg * 8,
                      &vbuf[buf][row * LDV_L + seg * 8]);
    }
  };

  // Q tile 16x64 -> two A operands (hd chunks of 32), resident in VGPRs
  union { v16bf v; v8bf hh[2]; } qa[2];
  {
    const bf16* qrow = Q + ((size_t)bh * S + q0 + l16) * 64;
#pragma unroll
    for (int c = 0; c < 2; ++c) {
      qa[c].hh[0] = *(const v8bf*)(qrow + c * 32 + half * 8);
      qa[c].hh[1] = *(const v8bf*)(qrow + c * 32 + 16 + half * 8);
    }
  }

  stage(0, 0);

  v8f acc0 = zero8(), acc1 = zero8(), acc2 = zero8(), acc3 = zero8();
  float mrow[8], lrow[8];
#pragma unroll
  for (int r = 0; r < 8; ++r) { mrow[r] = -1e30f; lrow[r] = 0.0f; }

  for (int kt = 0; kt < S; kt += 32) {
    const int buf = (kt >> 5) & 1;
    wait_async0();
    __syncthreads();                  // K/V tiles resident
    if (kt + 32 < S) stage(buf ^ 1, kt + 32);

    // ---- scores: S[16q x 32k] as two 16x16 C tiles, K over hd=64 ----
    v8f c0 = zero8(), c1 = zero8();
#pragma unroll
    for (int c = 0; c < 2; ++c) {
      union { v16bf v; v8bf hh[2]; } bk0, bk1;
      bk0.hh[0] = *(const v8bf*)&kbuf[buf][(l16)      * LDK_L + c * 32 + half * 8];
      bk0.hh[1] = *(const v8bf*)&kbuf[buf][(l16)      * LDK_L + c * 32 + 16 + half * 8];
      bk1.hh[0] = *(const v8bf*)&kbuf[buf][(l16 + 16) * LDK_L + c * 32 + half * 8];
      bk1.hh[1] = *(const v8bf*)&kbuf[buf][(l16 + 16) * LDK_L + c * 32 + 16 + half * 8];
      c0 = wmma_bf16(qa[c].v, bk0.v, c0);
      c1 = wmma_bf16(qa[c].v, bk1.v, c1);
    }

    // ---- online softmax (row = r + 8*half, spread over 16 lanes) ----
#pragma unroll
    for (int r = 0; r < 8; ++r) {
      float s0 = c0[r] * scale, s1 = c1[r] * scale;
      float v = fmaxf(s0, s1);
      v = fmaxf(v, __shfl_xor(v, 1));
      v = fmaxf(v, __shfl_xor(v, 2));
      v = fmaxf(v, __shfl_xor(v, 4));
      v = fmaxf(v, __shfl_xor(v, 8));
      float mn    = fmaxf(mrow[r], v);
      float alpha = __expf(mrow[r] - mn);
      mrow[r] = mn;
      float p0 = __expf(s0 - mn), p1 = __expf(s1 - mn);
      float rs = p0 + p1;
      rs += __shfl_xor(rs, 1);
      rs += __shfl_xor(rs, 2);
      rs += __shfl_xor(rs, 4);
      rs += __shfl_xor(rs, 8);
      lrow[r] = lrow[r] * alpha + rs;
      acc0[r] = acc0[r] * alpha;
      acc1[r] = acc1[r] * alpha;
      acc2[r] = acc2[r] * alpha;
      acc3[r] = acc3[r] * alpha;
      int row = r + half * 8;
      pbuf[wave][row][l16]      = to_bf16(p0);
      pbuf[wave][row][16 + l16] = to_bf16(p1);
    }
    __syncthreads();                  // C-layout -> A-layout bounce via LDS
    union { v16bf v; v8bf hh[2]; } pa;
    pa.hh[0] = *(const v8bf*)&pbuf[wave][l16][half * 8];
    pa.hh[1] = *(const v8bf*)&pbuf[wave][l16][16 + half * 8];
    __syncthreads();

    // ---- O += P(16x32) * V(32x64): 4 B operands from staged V^T ----
    union { v16bf v; v8bf hh[2]; } bv0, bv1, bv2, bv3;
    bv0.hh[0] = *(const v8bf*)&vbuf[buf][( 0 + l16) * LDV_L + half * 8];
    bv0.hh[1] = *(const v8bf*)&vbuf[buf][( 0 + l16) * LDV_L + 16 + half * 8];
    bv1.hh[0] = *(const v8bf*)&vbuf[buf][(16 + l16) * LDV_L + half * 8];
    bv1.hh[1] = *(const v8bf*)&vbuf[buf][(16 + l16) * LDV_L + 16 + half * 8];
    bv2.hh[0] = *(const v8bf*)&vbuf[buf][(32 + l16) * LDV_L + half * 8];
    bv2.hh[1] = *(const v8bf*)&vbuf[buf][(32 + l16) * LDV_L + 16 + half * 8];
    bv3.hh[0] = *(const v8bf*)&vbuf[buf][(48 + l16) * LDV_L + half * 8];
    bv3.hh[1] = *(const v8bf*)&vbuf[buf][(48 + l16) * LDV_L + 16 + half * 8];
    acc0 = wmma_bf16(pa.v, bv0.v, acc0);
    acc1 = wmma_bf16(pa.v, bv1.v, acc1);
    acc2 = wmma_bf16(pa.v, bv2.v, acc2);
    acc3 = wmma_bf16(pa.v, bv3.v, acc3);
  }

  // ---- finalize: O / l, write ctx [B,S,512] bf16 ----
#pragma unroll
  for (int r = 0; r < 8; ++r) {
    float rl = 1.0f / lrow[r];
    int srow = q0 + r + half * 8;
    bf16* crow = ctx + ((size_t)(b * S + srow)) * 512 + h * 64;
    crow[ 0 + l16] = to_bf16(acc0[r] * rl);
    crow[16 + l16] = to_bf16(acc1[r] * rl);
    crow[32 + l16] = to_bf16(acc2[r] * rl);
    crow[48 + l16] = to_bf16(acc3[r] * rl);
  }
}

// ---------------------------------------------------------------------------
// LayerNorm over N=512 columns, one row per 128-thread block.
// ---------------------------------------------------------------------------
__global__ __launch_bounds__(128) void k_layernorm(
    const float* __restrict__ in, const float* __restrict__ gamma,
    const float* __restrict__ beta, float* __restrict__ outF,
    bf16* __restrict__ outB, int N)
{
  const int row = blockIdx.x;
  const float* x = in + (size_t)row * N;
  __shared__ float red[4];
  const int tid = threadIdx.x;

  float s = 0.0f;
  for (int i = tid; i < N; i += 128) s += x[i];
  for (int m = 1; m < 32; m <<= 1) s += __shfl_xor(s, m);
  if ((tid & 31) == 0) red[tid >> 5] = s;
  __syncthreads();
  float mean = (red[0] + red[1] + red[2] + red[3]) / (float)N;
  __syncthreads();

  float v = 0.0f;
  for (int i = tid; i < N; i += 128) { float d = x[i] - mean; v += d * d; }
  for (int m = 1; m < 32; m <<= 1) v += __shfl_xor(v, m);
  if ((tid & 31) == 0) red[tid >> 5] = v;
  __syncthreads();
  float var  = (red[0] + red[1] + red[2] + red[3]) / (float)N;
  float rstd = rsqrtf(var + 1e-5f);

  for (int i = tid; i < N; i += 128) {
    float y = (x[i] - mean) * rstd * gamma[i] + beta[i];
    if (outF) outF[(size_t)row * N + i] = y;
    if (outB) outB[(size_t)row * N + i] = to_bf16(y);
  }
}

// ---------------------------------------------------------------------------
// Launcher
// ---------------------------------------------------------------------------
extern "C" void kernel_launch(void* const* d_in, const int* in_sizes, int n_in,
                              void* d_out, int out_size, void* d_ws, size_t ws_size,
                              hipStream_t stream)
{
  (void)in_sizes; (void)n_in; (void)out_size; (void)ws_size;

  const float* src = (const float*)d_in[0];
  const float* Wq  = (const float*)d_in[1];
  const float* Wk  = (const float*)d_in[2];
  const float* Wv  = (const float*)d_in[3];
  const float* Wo  = (const float*)d_in[4];
  const float* W1  = (const float*)d_in[5];
  const float* b1  = (const float*)d_in[6];
  const float* W2  = (const float*)d_in[7];
  const float* b2  = (const float*)d_in[8];
  const float* g1  = (const float*)d_in[9];
  const float* be1 = (const float*)d_in[10];
  const float* g2  = (const float*)d_in[11];
  const float* be2 = (const float*)d_in[12];

  const int B = 2, S = 4096, D = 512, H = 8, DFF = 2048;
  const int M = B * S;  // 8192

  char* ws = (char*)d_ws;
  const size_t MB = 1ull << 20;
  // --- scratch layout (overlayed; ~102 MB high-water) ---
  bf16*  WqB  = (bf16*)(ws + 0 * MB);
  bf16*  WkB  = (bf16*)(ws + 512 * 1024);
  bf16*  WvB  = (bf16*)(ws + 1 * MB);
  bf16*  WoB  = (bf16*)(ws + 1536 * 1024);
  bf16*  W1B  = (bf16*)(ws + 2 * MB);    // 2 MB
  bf16*  W2B  = (bf16*)(ws + 4 * MB);    // 2 MB
  bf16*  srcB = (bf16*)(ws + 6 * MB);    // 8 MB  (dead after QKV)
  float* qf   = (float*)(ws + 14 * MB);  // 16 MB (dead after RoPE)
  float* kf   = (float*)(ws + 30 * MB);  // 16 MB (dead after RoPE)
  bf16*  vb   = (bf16*)(ws + 46 * MB);   // 8 MB  (dead after transpose)
  bf16*  qb   = (bf16*)(ws + 54 * MB);   // 8 MB
  bf16*  kb   = (bf16*)(ws + 62 * MB);   // 8 MB
  bf16*  vtb  = (bf16*)(ws + 70 * MB);   // 8 MB
  bf16*  ctxB = (bf16*)(ws + 6 * MB);    // reuse srcB
  float* y1   = (float*)(ws + 14 * MB);  // reuse qf
  float* xf   = (float*)(ws + 30 * MB);  // reuse kf
  bf16*  xB   = (bf16*)(ws + 46 * MB);   // reuse vb
  bf16*  hB   = (bf16*)(ws + 54 * MB);   // 32 MB, reuse qb/kb/vtb
  float* y2   = (float*)(ws + 86 * MB);  // 16 MB

  auto cvt = [&](const float* in, bf16* out, int n) {
    int n4 = n / 4;
    k_cvt<<<(n4 + 255) / 256, 256, 0, stream>>>(in, out, n4);
  };
  cvt(src, srcB, M * D);
  cvt(Wq, WqB, D * D);
  cvt(Wk, WkB, D * D);
  cvt(Wv, WvB, D * D);
  cvt(Wo, WoB, D * D);
  cvt(W1, W1B, DFF * D);
  cvt(W2, W2B, D * DFF);

  const dim3 blk(256);
  // QKV projections (q,k -> fp32 for RoPE; v -> bf16 then transpose)
  k_gemm<<<dim3(D / 128, M / 64), blk, 0, stream>>>(srcB, WqB, nullptr, nullptr, qf, nullptr, M, D, D, 0);
  k_gemm<<<dim3(D / 128, M / 64), blk, 0, stream>>>(srcB, WkB, nullptr, nullptr, kf, nullptr, M, D, D, 0);
  k_gemm<<<dim3(D / 128, M / 64), blk, 0, stream>>>(srcB, WvB, nullptr, nullptr, nullptr, vb, M, D, D, 0);

  const int nr = B * S * H * 32;
  k_rope<<<(nr + 255) / 256, 256, 0, stream>>>(qf, qb, B * S * H, S, H);
  k_rope<<<(nr + 255) / 256, 256, 0, stream>>>(kf, kb, B * S * H, S, H);

  const size_t nt = (size_t)M * D;
  k_transpose_v<<<(unsigned)((nt + 255) / 256), 256, 0, stream>>>(vb, vtb, S, H);

  k_flash<<<dim3(S / 64, B * H), dim3(128), 0, stream>>>(qb, kb, vtb, ctxB, S, H);

  // Wo projection + residual(src), then LN1 -> x (fp32 + bf16)
  k_gemm<<<dim3(D / 128, M / 64), blk, 0, stream>>>(ctxB, WoB, nullptr, src, y1, nullptr, M, D, D, 0);
  k_layernorm<<<M, 128, 0, stream>>>(y1, g1, be1, xf, xB, D);

  // FFN
  k_gemm<<<dim3(DFF / 128, M / 64), blk, 0, stream>>>(xB, W1B, b1, nullptr, nullptr, hB, M, DFF, D, 1);
  k_gemm<<<dim3(D / 128, M / 64), blk, 0, stream>>>(hB, W2B, b2, xf, y2, nullptr, M, D, DFF, 0);
  k_layernorm<<<M, 128, 0, stream>>>(y2, g2, be2, (float*)d_out, nullptr, D);
}
